// NeuralMemory_73959336837216
// MI455X (gfx1250) — matrix operand order repaired and checked by reading, not compiled, not run on hardware
//
#include <hip/hip_runtime.h>
#include <hip/hip_bf16.h>

// ---------------- problem constants ----------------
#define BB   2
#define SS   2048
#define DD   512
#define HIDN 1024
#define MMETA 64
#define TT   (MMETA + SS)     // 2112 tokens per batch
#define NTOK (BB * TT)        // 4224 rows
#define NHD  8
#define DHD  64
#define WINA 256
#define KC   4
#define TPAD (TT + 16)        // padded key stride for vT (attention pad tile)

typedef __attribute__((ext_vector_type(16))) _Float16 v16h;
typedef __attribute__((ext_vector_type(8)))  float    v8f;

union Frag { unsigned u[8]; v16h h; };

// A-fragment K offset for 16-bit 16x32 A tile (contiguous: k = 16*hi + 2*j, matching B)
__device__ __forceinline__ int a_koff(int j, int hi) {
    return 2 * j + 16 * hi;
}

// ---------------- weight transpose + f16 convert: dst[n*K+k] = src[k*N+n] ----
__global__ void transpose_f16(const float* __restrict__ src, _Float16* __restrict__ dst,
                              int K, int Nc) {
    size_t idx = (size_t)blockIdx.x * 256 + threadIdx.x;
    if (idx >= (size_t)K * Nc) return;
    int k = (int)(idx / Nc), n = (int)(idx % Nc);
    dst[(size_t)n * K + k] = (_Float16)src[idx];
}

// ---------------- build xm (meta ++ x) as f16 [NTOK x DD] --------------------
__global__ void build_xm(const float* __restrict__ x, const float* __restrict__ meta,
                         _Float16* __restrict__ xm) {
    size_t idx = (size_t)blockIdx.x * 256 + threadIdx.x;
    if (idx >= (size_t)NTOK * DD) return;
    int row = (int)(idx / DD), d = (int)(idx % DD);
    int b = row / TT, t = row % TT;
    float v = (t < MMETA) ? meta[(size_t)t * DD + d]
                          : x[((size_t)b * SS + (t - MMETA)) * DD + d];
    xm[idx] = (_Float16)v;
}

// ---------------- depthwise causal conv (K=4) + L2 normalize, f16 out --------
__global__ void conv_norm(const float* __restrict__ qlin, const float* __restrict__ kern,
                          _Float16* __restrict__ qn) {
    int row = blockIdx.x;                 // token
    int b = row / TT, t = row % TT;
    int d0 = threadIdx.x;                 // 256 threads, 2 channels each
    float v[2];
#pragma unroll
    for (int i = 0; i < 2; ++i) {
        int d = d0 + i * 256;
        float acc = 0.f;
#pragma unroll
        for (int kk = 0; kk < KC; ++kk) {
            int ts = t + kk - (KC - 1);
            if (ts >= 0) acc += qlin[((size_t)b * TT + ts) * DD + d] * kern[kk * DD + d];
        }
        v[i] = acc;
    }
    __shared__ float red[256];
    __shared__ float invn;
    red[d0] = v[0] * v[0] + v[1] * v[1];
    __syncthreads();
    for (int s = 128; s > 0; s >>= 1) {
        if (d0 < s) red[d0] += red[d0 + s];
        __syncthreads();
    }
    if (d0 == 0) invn = 1.f / fmaxf(sqrtf(red[0]), 1e-12f);
    __syncthreads();
#pragma unroll
    for (int i = 0; i < 2; ++i)
        qn[(size_t)row * DD + d0 + i * 256] = (_Float16)(v[i] * invn);
}

// ---------------- generic WMMA GEMM: C = post(A @ Wt^T + bias) ---------------
// A: [Mrows x Kd] f16 row-major; Wt: [Ncols x Kd] f16 (pre-transposed weight).
// Block = 128 thr (4 waves); block tile 64x64 of C; wave tile 16x64.
template<bool SILU, bool RESID, bool WF32, bool WF16, bool WVT, bool WFIN>
__global__ void gemm_wmma(const _Float16* __restrict__ A, const _Float16* __restrict__ Wt,
                          const float* __restrict__ bias, const float* __restrict__ resid,
                          float* __restrict__ outF, _Float16* __restrict__ outH,
                          int Kd, int Ncols) {
    const int lane = threadIdx.x & 31;
    const int wav  = threadIdx.x >> 5;
    const int m0 = blockIdx.y * 64 + wav * 16;
    const int n0 = blockIdx.x * 64;
    const int m  = lane & 15;
    const int hi = lane >> 4;

    v8f acc[4] = {};
    const _Float16* arow = A + (size_t)(m0 + m) * Kd;

    for (int k0 = 0; k0 < Kd; k0 += 32) {
        Frag a;
#pragma unroll
        for (int j = 0; j < 8; ++j)
            a.u[j] = *(const unsigned*)(arow + k0 + a_koff(j, hi));
#pragma unroll
        for (int t = 0; t < 4; ++t) {
            const _Float16* wrow = Wt + (size_t)(n0 + t * 16 + m) * Kd + k0;
            Frag bw;
#pragma unroll
            for (int j = 0; j < 8; ++j)
                bw.u[j] = *(const unsigned*)(wrow + 2 * j + 16 * hi);
            acc[t] = __builtin_amdgcn_wmma_f32_16x16x32_f16(
                false, a.h, false, bw.h, (short)0, acc[t], false, false);
        }
    }

#pragma unroll
    for (int t = 0; t < 4; ++t) {
        int col = n0 + 16 * t + m;
        float bvc = bias[col];
#pragma unroll
        for (int r = 0; r < 8; ++r) {
            int row = m0 + 8 * hi + r;
            float v = acc[t][r] + bvc;
            if (SILU)  v = v / (1.f + __expf(-v));          // x * sigmoid(x)
            if (RESID) v += resid[(size_t)row * Ncols + col];
            if (WF32)  outF[(size_t)row * Ncols + col] = v;
            if (WF16)  outH[(size_t)row * Ncols + col] = (_Float16)v;
            if (WVT) {   // transposed per-batch V layout: vT[b][col][t], stride TPAD
                int b2 = row / TT, t2 = row % TT;
                outH[((size_t)(b2 * Ncols + col)) * TPAD + t2] = (_Float16)v;
            }
            if (WFIN) {  // final projection: drop meta rows, write [B,S,D] fp32
                int b2 = row / TT, t2 = row % TT;
                if (t2 >= MMETA)
                    outF[((size_t)(b2 * SS + (t2 - MMETA))) * Ncols + col] = v;
            }
        }
    }
}

// ---------------- sliding-window attention (one wave per b,h,qtile) ----------
__global__ void attn_swa(const _Float16* __restrict__ qh, const _Float16* __restrict__ kh,
                         const _Float16* __restrict__ vT, _Float16* __restrict__ oh) {
    const int lane = threadIdx.x & 31;
    const int q0 = blockIdx.x * 16;
    const int h  = blockIdx.y;
    const int b  = blockIdx.z;
    const int m  = lane & 15;
    const int hi = lane >> 4;

    __shared__ float    sc[16][288];
    __shared__ _Float16 pb[16][288];
    __shared__ float    rs[16];

    // Q fragments: two K=32 chunks over DH=64
    Frag aq[2];
    const _Float16* qrow = qh + ((size_t)(b * TT + q0 + m)) * DD + h * DHD;
#pragma unroll
    for (int c = 0; c < 2; ++c)
#pragma unroll
        for (int j = 0; j < 8; ++j)
            aq[c].u[j] = *(const unsigned*)(qrow + 32 * c + a_koff(j, hi));

    int kb = q0 - (WINA - 1); if (kb < 0) kb = 0; kb &= ~15;
    int ntiles = (q0 + 16 - kb) >> 4;      // <= 17
    const float scale = 0.125f;            // 1/sqrt(64)

    for (int ti = 0; ti < ntiles; ++ti) {
        int kt = kb + ti * 16;
        v8f s = {};
        const _Float16* krow = kh + ((size_t)(b * TT + kt + m)) * DD + h * DHD;
#pragma unroll
        for (int c = 0; c < 2; ++c) {
            Frag bk;
#pragma unroll
            for (int j = 0; j < 8; ++j)
                bk.u[j] = *(const unsigned*)(krow + 32 * c + 2 * j + 16 * hi);
            s = __builtin_amdgcn_wmma_f32_16x16x32_f16(
                false, aq[c].h, false, bk.h, (short)0, s, false, false);
        }
#pragma unroll
        for (int r = 0; r < 8; ++r) {
            int rl = 8 * hi + r;
            int qA = q0 + rl, kA = kt + m;
            float v = s[r] * scale;
            bool ok = (kA <= qA) && (qA - kA < WINA);
            sc[rl][ti * 16 + m] = ok ? v : -1e9f;
        }
    }
    __syncthreads();

    int nk  = ntiles * 16;
    int npt = (ntiles + 1) & ~1;           // pad tile count to even (K=32 chunks)
    int nkp = npt * 16;
    // zero the probability pad columns
    for (int k = nk + lane; k < nkp; k += 32)
        for (int r2 = 0; r2 < 16; ++r2) pb[r2][k] = (_Float16)0.f;

    if (lane < 16) {                       // exact softmax per query row
        int row = lane;
        float mx = -1e30f;
        for (int k = 0; k < nk; ++k) mx = fmaxf(mx, sc[row][k]);
        float sum = 0.f;
        for (int k = 0; k < nk; ++k) {
            float e = __expf(sc[row][k] - mx);
            sum += e;
            pb[row][k] = (_Float16)e;
        }
        rs[row] = sum;
    }
    __syncthreads();

    v8f o[4] = {};
    int nch = npt >> 1;
    for (int c = 0; c < nch; ++c) {
        Frag ap;
#pragma unroll
        for (int j = 0; j < 8; ++j)
            ap.u[j] = *(const unsigned*)&pb[m][32 * c + a_koff(j, hi)];
#pragma unroll
        for (int jt = 0; jt < 4; ++jt) {
            Frag bvf;
            const _Float16* vrow = vT + ((size_t)(b * DD + h * DHD + jt * 16 + m)) * TPAD
                                      + kb + 32 * c;
#pragma unroll
            for (int j = 0; j < 8; ++j)
                bvf.u[j] = *(const unsigned*)(vrow + 2 * j + 16 * hi);
            o[jt] = __builtin_amdgcn_wmma_f32_16x16x32_f16(
                false, ap.h, false, bvf.h, (short)0, o[jt], false, false);
        }
    }
#pragma unroll
    for (int jt = 0; jt < 4; ++jt)
#pragma unroll
        for (int r = 0; r < 8; ++r) {
            int rl = 8 * hi + r;
            float inv = 1.f / rs[rl];
            oh[((size_t)(b * TT + q0 + rl)) * DD + h * DHD + jt * 16 + m] =
                (_Float16)(o[jt][r] * inv);
        }
}

// ---------------- host orchestration ----------------------------------------
extern "C" void kernel_launch(void* const* d_in, const int* in_sizes, int n_in,
                              void* d_out, int out_size, void* d_ws, size_t ws_size,
                              hipStream_t stream) {
    (void)in_sizes; (void)n_in; (void)out_size; (void)ws_size;
    const float* x       = (const float*)d_in[0];
    const float* meta    = (const float*)d_in[1];
    const float* qp_w    = (const float*)d_in[2];
    const float* qp_b    = (const float*)d_in[3];
    const float* qp_conv = (const float*)d_in[4];
    const float* w_in    = (const float*)d_in[5];
    const float* b_in    = (const float*)d_in[6];
    const float* w_hid   = (const float*)d_in[7];
    const float* b_hid   = (const float*)d_in[8];
    const float* w_out   = (const float*)d_in[9];
    const float* b_out   = (const float*)d_in[10];
    const float* wq      = (const float*)d_in[11];
    const float* bq      = (const float*)d_in[12];
    const float* wk      = (const float*)d_in[13];
    const float* bk2     = (const float*)d_in[14];
    const float* wv      = (const float*)d_in[15];
    const float* bv2     = (const float*)d_in[16];
    const float* wo      = (const float*)d_in[17];
    const float* bo      = (const float*)d_in[18];

    char* ws = (char*)d_ws;
    size_t off = 0;
    auto alloc = [&](size_t bytes) {
        size_t o = off; off += (bytes + 255) & ~(size_t)255; return o;
    };
    _Float16* xm   = (_Float16*)(ws + alloc((size_t)NTOK * DD * 2));
    float*    qlin = (float*)   (ws + alloc((size_t)NTOK * DD * 4));
    _Float16* qn   = (_Float16*)(ws + alloc((size_t)NTOK * DD * 2));
    float*    hAf  = (float*)   (ws + alloc((size_t)NTOK * HIDN * 4));
    _Float16* hAh  = (_Float16*)(ws + alloc((size_t)NTOK * HIDN * 2));
    float*    hBf  = (float*)   (ws + alloc((size_t)NTOK * HIDN * 4));
    _Float16* hBh  = (_Float16*)(ws + alloc((size_t)NTOK * HIDN * 2));
    _Float16* rh   = (_Float16*)(ws + alloc((size_t)NTOK * DD * 2));
    _Float16* qhh  = (_Float16*)(ws + alloc((size_t)NTOK * DD * 2));
    _Float16* khh  = (_Float16*)(ws + alloc((size_t)NTOK * DD * 2));
    _Float16* vth  = (_Float16*)(ws + alloc((size_t)BB * DD * TPAD * 2));
    _Float16* ohh  = (_Float16*)(ws + alloc((size_t)NTOK * DD * 2));
    _Float16* qpwT = (_Float16*)(ws + alloc((size_t)DD * DD * 2));
    _Float16* winT = (_Float16*)(ws + alloc((size_t)DD * HIDN * 2));
    _Float16* whT0 = (_Float16*)(ws + alloc((size_t)HIDN * HIDN * 2));
    _Float16* whT1 = (_Float16*)(ws + alloc((size_t)HIDN * HIDN * 2));
    _Float16* woutT= (_Float16*)(ws + alloc((size_t)HIDN * DD * 2));
    _Float16* wqT  = (_Float16*)(ws + alloc((size_t)DD * DD * 2));
    _Float16* wkT  = (_Float16*)(ws + alloc((size_t)DD * DD * 2));
    _Float16* wvT  = (_Float16*)(ws + alloc((size_t)DD * DD * 2));
    _Float16* woT  = (_Float16*)(ws + alloc((size_t)DD * DD * 2));

    auto tr = [&](const float* s, _Float16* d, int K, int Nc) {
        size_t tot = (size_t)K * Nc;
        transpose_f16<<<dim3((unsigned)((tot + 255) / 256)), dim3(256), 0, stream>>>(s, d, K, Nc);
    };
    tr(qp_w, qpwT, DD, DD);
    tr(w_in, winT, DD, HIDN);
    tr(w_hid, whT0, HIDN, HIDN);
    tr(w_hid + (size_t)HIDN * HIDN, whT1, HIDN, HIDN);
    tr(w_out, woutT, HIDN, DD);
    tr(wq, wqT, DD, DD);
    tr(wk, wkT, DD, DD);
    tr(wv, wvT, DD, DD);
    tr(wo, woT, DD, DD);

    {
        size_t tot = (size_t)NTOK * DD;
        build_xm<<<dim3((unsigned)((tot + 255) / 256)), dim3(256), 0, stream>>>(x, meta, xm);
    }

    const dim3 blk(128);
    const dim3 gD(DD / 64, NTOK / 64);     // 8 x 66
    const dim3 gH(HIDN / 64, NTOK / 64);   // 16 x 66

    // 1) q linear
    gemm_wmma<false,false,true,false,false,false><<<gD, blk, 0, stream>>>(
        xm, qpwT, qp_b, nullptr, qlin, nullptr, DD, DD);
    // 2) causal depthwise conv + l2 norm
    conv_norm<<<dim3(NTOK), dim3(256), 0, stream>>>(qlin, qp_conv, qn);
    // 3) mlp in: h = silu(q @ w_in + b)
    gemm_wmma<true,false,true,true,false,false><<<gH, blk, 0, stream>>>(
        qn, winT, b_in, nullptr, hAf, hAh, DD, HIDN);
    // 4) residual hidden layers
    gemm_wmma<true,true,true,true,false,false><<<gH, blk, 0, stream>>>(
        hAh, whT0, b_hid, hAf, hBf, hBh, HIDN, HIDN);
    gemm_wmma<true,true,true,true,false,false><<<gH, blk, 0, stream>>>(
        hBh, whT1, b_hid + HIDN, hBf, hAf, hAh, HIDN, HIDN);
    // 5) mlp out: r = silu(h @ w_out + b)
    gemm_wmma<true,false,false,true,false,false><<<gD, blk, 0, stream>>>(
        hAh, woutT, b_out, nullptr, nullptr, rh, HIDN, DD);
    // 6) q/k/v projections (v written transposed for attention PV fragments)
    gemm_wmma<false,false,false,true,false,false><<<gD, blk, 0, stream>>>(
        rh, wqT, bq, nullptr, nullptr, qhh, DD, DD);
    gemm_wmma<false,false,false,true,false,false><<<gD, blk, 0, stream>>>(
        rh, wkT, bk2, nullptr, nullptr, khh, DD, DD);
    gemm_wmma<false,false,false,false,true,false><<<gD, blk, 0, stream>>>(
        rh, wvT, bv2, nullptr, nullptr, vth, DD, DD);
    // 7) sliding-window attention
    attn_swa<<<dim3(TT / 16, NHD, BB), dim3(32), 0, stream>>>(qhh, khh, vth, ohh);
    // 8) output projection, slice off meta rows into d_out [B,S,D] fp32
    gemm_wmma<false,false,false,false,false,true><<<gD, blk, 0, stream>>>(
        ohh, woT, bo, nullptr, (float*)d_out, nullptr, DD, DD);
}
